// tPatchGNN_9457517985846
// MI455X (gfx1250) — matrix operand order repaired and checked by reading, not compiled, not run on hardware
//
#include <hip/hip_runtime.h>
#include <hip/hip_bf16.h>

// ---- problem constants ----
#define kB 8
#define kM 16
#define kL 24
#define kN 50
#define kLP 32
#define kHID 64
#define kTE 10
#define kNHEAD 4
#define kNLAYER 2
#define kND 16
#define kDFF 2048
#define kIN 11
#define kTTCN 63
#define kSEQ (kB * kN * kM)      // 6400 sequences / tokens
#define kDTOK (kB * kN * kLP)    // 12800 decoder tokens

typedef __attribute__((ext_vector_type(16))) _Float16 v16h;
typedef __attribute__((ext_vector_type(8)))  float    v8f;

// ----------------------------------------------------------------------
// Weight prep: f32 -> f16 with optional zero padding (rows_src x k_src)
// into (rows_dst x k_dst). Makes every GEMM inner loop guard-free.
// ----------------------------------------------------------------------
__global__ void pad_cvt_kernel(const float* __restrict__ src, _Float16* __restrict__ dst,
                               int rs, int ks, int rd, int kd) {
    int i = blockIdx.x * blockDim.x + threadIdx.x;
    if (i >= rd * kd) return;
    int r = i / kd, k = i % kd;
    dst[i] = (r < rs && k < ks) ? (_Float16)src[(long)r * ks + k] : (_Float16)0.f;
}

// ----------------------------------------------------------------------
// Wave32 WMMA GEMM: C[m,n] = act( sum_k A[m,k] * W[n,k] + bias[n] )
// A: (M x K) f32 row-major (16B aligned, K%32==0). W: (N x K) f16 padded.
// One wave per block; register tile = MT m-tiles x NT n-tiles, so each
// B fragment feeds MT wmmas and each A fragment feeds NT wmmas
// (MT*NT wmmas per 20-load k-step -> latency-tolerant, less L2 traffic).
// Requires M % (16*MT) == 0, N % (16*NT) == 0, K % 32 == 0.
// Fragment maps (A/B: contiguous k, same in both; C/D per CDNA5 ISA 7.12.2):
//   A f16 16x32: K = h + (lane>=16)*16           -> one 16-float run
//   B f16 32x16: K = h + (lane>=16)*16           -> one 16-half run
//   C/D f32:     m = r + (lane>=16)*8, n = lane&15
// ----------------------------------------------------------------------
template <int ACT, int MT, int NT>  // ACT: 0=none, 1=relu
__global__ void gemm_nt(const float* __restrict__ A, const _Float16* __restrict__ W,
                        const float* __restrict__ bias, float* __restrict__ C,
                        int Ncols, int K) {
    const int lane = threadIdx.x & 31;
    const int m0 = blockIdx.x * 16 * MT;
    const int n0 = blockIdx.y * 16 * NT;
    const int abase = (lane & 16) ? 16 : 0;
    const int bbase = (lane & 16) ? 16 : 0;

    const _Float16* Wrow[NT];
    float bv[NT];
#pragma unroll
    for (int j = 0; j < NT; ++j) {
        int nc = n0 + j * 16 + (lane & 15);
        Wrow[j] = W + (long)nc * K;
        bv[j] = bias[nc];
    }

    v8f acc[MT][NT] = {};
#pragma unroll 2
    for (int k0 = 0; k0 < K; k0 += 32) {
        v16h bf[NT];
#pragma unroll
        for (int j = 0; j < NT; ++j) bf[j] = *(const v16h*)(Wrow[j] + k0 + bbase);
#pragma unroll
        for (int i = 0; i < MT; ++i) {
            const float* Ar = A + (long)(m0 + i * 16 + (lane & 15)) * K + k0 + abase;
            float4 a0 = *(const float4*)(Ar);
            float4 a1 = *(const float4*)(Ar + 4);
            float4 a2 = *(const float4*)(Ar + 8);
            float4 a3 = *(const float4*)(Ar + 12);
            v16h af;
            af[0] = (_Float16)a0.x;  af[1] = (_Float16)a0.y;
            af[2] = (_Float16)a0.z;  af[3] = (_Float16)a0.w;
            af[4] = (_Float16)a1.x;  af[5] = (_Float16)a1.y;
            af[6] = (_Float16)a1.z;  af[7] = (_Float16)a1.w;
            af[8] = (_Float16)a2.x;  af[9] = (_Float16)a2.y;
            af[10] = (_Float16)a2.z; af[11] = (_Float16)a2.w;
            af[12] = (_Float16)a3.x; af[13] = (_Float16)a3.y;
            af[14] = (_Float16)a3.z; af[15] = (_Float16)a3.w;
#pragma unroll
            for (int j = 0; j < NT; ++j)
                acc[i][j] = __builtin_amdgcn_wmma_f32_16x16x32_f16(
                    false, af, false, bf[j], (short)0, acc[i][j], false, false);
        }
    }
#pragma unroll
    for (int i = 0; i < MT; ++i) {
        int mbase = m0 + i * 16 + ((lane & 16) ? 8 : 0);
#pragma unroll
        for (int j = 0; j < NT; ++j) {
            int nc = n0 + j * 16 + (lane & 15);
#pragma unroll
            for (int r = 0; r < 8; ++r) {
                float v = acc[i][j][r] + bv[j];
                if (ACT == 1) v = fmaxf(v, 0.f);
                C[(long)(mbase + r) * Ncols + nc] = v;
            }
        }
    }
}

// ----------------------------------------------------------------------
// Fused TTCN stage: TE + fg1 + fg2 (VALU in LDS) + fg3 (WMMA vs padded
// f16 weights; 24x693 logits live only in LDS -> the 426MB filt tensor
// never reaches memory) + masked softmax over L + contraction with x_int.
// One block = one sequence.
// ----------------------------------------------------------------------
__global__ void ttcn_kernel(const float* __restrict__ X, const float* __restrict__ TTS,
                            const float* __restrict__ MSK,
                            const float* __restrict__ tesw, const float* __restrict__ tesb,
                            const float* __restrict__ tepw, const float* __restrict__ tepb,
                            const float* __restrict__ fg1w, const float* __restrict__ fg1b,
                            const float* __restrict__ fg2w, const float* __restrict__ fg2b,
                            const _Float16* __restrict__ fg3h,  // padded 704 x 64 f16
                            const float* __restrict__ fg3b,
                            const float* __restrict__ Tbias, float* __restrict__ Xout) {
    __shared__ float    s_xint[kL][12];
    __shared__ float    s_m[kL];
    __shared__ float    s_h1[kL][64];
    __shared__ float    s_h2[32][64];     // zero-padded rows/cols for WMMA
    __shared__ _Float16 s_filt[kL][704];  // 24 x 693 logits (padded to 704)
    __shared__ float    s_fb[704];        // fg3 bias staged in LDS
    __shared__ float    s_out[64];

    const int s   = blockIdx.x;
    const int b   = s / (kN * kM);
    const int n   = (s / kM) % kN;
    const int mp  = s % kM;
    const int tid = threadIdx.x;

    // ---- stage fg3 bias ----
    for (int i = tid; i < 704; i += blockDim.x) s_fb[i] = (i < 693) ? fg3b[i] : 0.f;

    // ---- inputs + time embedding -> x_int (24 x 11) ----
    for (int i = tid; i < kL * 12; i += blockDim.x) {
        int l = i / 12, d = i % 12;
        long idx = (((long)b * kM + mp) * kL + l) * kN + n;
        float tt = TTS[idx];
        float v;
        if (d == 0)      v = X[idx];
        else if (d == 1) v = tt * tesw[0] + tesb[0];
        else if (d < 11) v = __sinf(tt * tepw[d - 2] + tepb[d - 2]);
        else             v = 0.f;
        s_xint[l][d] = v;
        if (d == 0) s_m[l] = MSK[idx];
    }
    __syncthreads();

    // ---- h1 = relu(x_int @ fg1^T + b)  (24 x 63) ----
    for (int i = tid; i < kL * 64; i += blockDim.x) {
        int l = i / 64, j = i % 64;
        float v = 0.f;
        if (j < kTTCN) {
            v = fg1b[j];
            for (int d = 0; d < kIN; ++d) v += s_xint[l][d] * fg1w[j * kIN + d];
            v = fmaxf(v, 0.f);
        }
        s_h1[l][j] = v;
    }
    __syncthreads();

    // ---- h2 = relu(h1 @ fg2^T + b), zero-padded to 32x64 for WMMA ----
    for (int i = tid; i < 32 * 64; i += blockDim.x) {
        int l = i / 64, j = i % 64;
        float v = 0.f;
        if (l < kL && j < kTTCN) {
            v = fg2b[j];
            for (int d = 0; d < kTTCN; ++d) v += s_h1[l][d] * fg2w[j * kTTCN + d];
            v = fmaxf(v, 0.f);
        }
        s_h2[l][j] = v;
    }
    __syncthreads();

    // ---- fg3 via WMMA: filt(24x693) = h2(24x63) @ fg3^T + bias ----
    {
        const int wave = tid >> 5, lane = tid & 31;
        const int abase = (lane & 16) ? 16 : 0;
        const int bbase = (lane & 16) ? 16 : 0;
        for (int job = wave; job < 2 * 44; job += 8) {  // 2 m-tiles x 44 n-tiles
            const int mt = job & 1, nt = job >> 1;
            const int ncol = nt * 16 + (lane & 15);
            const float* hr = &s_h2[mt * 16 + (lane & 15)][0];
            const _Float16* Wrow = fg3h + (long)ncol * 64;
            v8f acc = {};
#pragma unroll
            for (int k0 = 0; k0 < 64; k0 += 32) {
                v16h bf = *(const v16h*)(Wrow + k0 + bbase);
                float4 a0 = *(const float4*)(hr + k0 + abase);
                float4 a1 = *(const float4*)(hr + k0 + abase + 4);
                float4 a2 = *(const float4*)(hr + k0 + abase + 8);
                float4 a3 = *(const float4*)(hr + k0 + abase + 12);
                v16h af;
                af[0] = (_Float16)a0.x;  af[1] = (_Float16)a0.y;
                af[2] = (_Float16)a0.z;  af[3] = (_Float16)a0.w;
                af[4] = (_Float16)a1.x;  af[5] = (_Float16)a1.y;
                af[6] = (_Float16)a1.z;  af[7] = (_Float16)a1.w;
                af[8] = (_Float16)a2.x;  af[9] = (_Float16)a2.y;
                af[10] = (_Float16)a2.z; af[11] = (_Float16)a2.w;
                af[12] = (_Float16)a3.x; af[13] = (_Float16)a3.y;
                af[14] = (_Float16)a3.z; af[15] = (_Float16)a3.w;
                acc = __builtin_amdgcn_wmma_f32_16x16x32_f16(false, af, false, bf, (short)0,
                                                             acc, false, false);
            }
            if (ncol < 693) {
                float bb = s_fb[ncol];
#pragma unroll
                for (int r = 0; r < 8; ++r) {
                    int mm = mt * 16 + r + ((lane & 16) ? 8 : 0);
                    if (mm < kL) s_filt[mm][ncol] = (_Float16)(acc[r] + bb);
                }
            }
        }
    }
    if (tid < 64) s_out[tid] = 0.f;
    __syncthreads();

    // ---- masked softmax over L per channel + contraction with x_int ----
    for (int c = tid; c < 693; c += blockDim.x) {
        const int k = c / kIN, d = c % kIN;  // channel -> (filter k, input dim d)
        float mx = -1e30f;
        for (int l = 0; l < kL; ++l) {
            float mm = s_m[l];
            float f = (float)s_filt[l][c] * mm + (1.f - mm) * (-1e8f);
            mx = fmaxf(mx, f);
        }
        float ssum = 0.f, numer = 0.f;
        for (int l = 0; l < kL; ++l) {
            float mm = s_m[l];
            float f = (float)s_filt[l][c] * mm + (1.f - mm) * (-1e8f);
            float e = __expf(f - mx);
            ssum += e;
            numer += e * s_xint[l][d];
        }
        atomicAdd(&s_out[k], numer / ssum);
    }
    __syncthreads();

    // ---- h_t = relu(ttcn + T_bias); concat mask_patch -> x0 (seq, 64) ----
    if (tid < 64) {
        float v;
        if (tid < kTTCN) {
            v = fmaxf(s_out[tid] + Tbias[tid], 0.f);
        } else {
            float sm = 0.f;
            for (int l = 0; l < kL; ++l) sm += s_m[l];
            v = (sm > 0.f) ? 1.f : 0.f;
        }
        Xout[(long)s * kHID + tid] = v;
    }
}

// ---- x + positional encoding (pos = patch index within sequence of 16) ----
__global__ void addpe_kernel(const float* __restrict__ x, float* __restrict__ xpe) {
    int i = blockIdx.x * blockDim.x + threadIdx.x;
    if (i >= kSEQ * kHID) return;
    int d = i & 63;
    int pos = (i >> 6) & (kM - 1);
    float div = __expf(-(float)(d & ~1) * (9.210340371976184f / 64.f));  // ln(10000)/64
    float ang = (float)pos * div;
    float pe = (d & 1) ? __cosf(ang) : __sinf(ang);
    xpe[i] = x[i] + pe;
}

// ---- 4-head attention, one block per (sequence, head), S=16, dh=16 ----
__global__ void attn_kernel(const float* __restrict__ qkv, float* __restrict__ o) {
    const int seq = blockIdx.x >> 2;
    const int h = blockIdx.x & 3;
    __shared__ float sq[16][16], sk[16][16], sv[16][16], satt[16][16];
    __shared__ float smx[16], ssum[16];
    const int tid = threadIdx.x;
    const int i = tid >> 4, j = tid & 15;
    long base = ((long)seq * 16 + i) * 192 + h * 16 + j;
    sq[i][j] = qkv[base];
    sk[i][j] = qkv[base + 64];
    sv[i][j] = qkv[base + 128];
    __syncthreads();
    float sc = 0.f;
    for (int d = 0; d < 16; ++d) sc += sq[i][d] * sk[j][d];
    satt[i][j] = sc * 0.25f;  // 1/sqrt(16)
    __syncthreads();
    if (j == 0) {
        float mx = satt[i][0];
        for (int d = 1; d < 16; ++d) mx = fmaxf(mx, satt[i][d]);
        smx[i] = mx;
    }
    __syncthreads();
    float e = __expf(satt[i][j] - smx[i]);
    __syncthreads();
    satt[i][j] = e;
    __syncthreads();
    if (j == 0) {
        float s2 = 0.f;
        for (int d = 0; d < 16; ++d) s2 += satt[i][d];
        ssum[i] = s2;
    }
    __syncthreads();
    float acc = 0.f;
    for (int d = 0; d < 16; ++d) acc += satt[i][d] * sv[d][j];
    o[((long)seq * 16 + i) * 64 + h * 16 + j] = acc / ssum[i];
}

// ---- out = LayerNorm(a + b) * g + beta, one block (64 thr) per token ----
__global__ void addln_kernel(const float* __restrict__ a, const float* __restrict__ bsrc,
                             const float* __restrict__ g, const float* __restrict__ be,
                             float* __restrict__ out) {
    const int t = blockIdx.x, d = threadIdx.x;
    __shared__ float red[64];
    float v = a[(long)t * 64 + d] + bsrc[(long)t * 64 + d];
    red[d] = v;
    __syncthreads();
    for (int s2 = 32; s2 > 0; s2 >>= 1) {
        if (d < s2) red[d] += red[d + s2];
        __syncthreads();
    }
    float mu = red[0] * (1.f / 64.f);
    __syncthreads();
    float dv = v - mu;
    red[d] = dv * dv;
    __syncthreads();
    for (int s2 = 32; s2 > 0; s2 >>= 1) {
        if (d < s2) red[d] += red[d + s2];
        __syncthreads();
    }
    float var = red[0] * (1.f / 64.f);
    out[(long)t * 64 + d] = dv * rsqrtf(var + 1e-5f) * g[d] + be[d];
}

// ---- node gates: g1/g2 (tanh-relu) and xp1/xp2 = gate * nvl(x) ----
__global__ void gate_kernel(const float* __restrict__ x, const float* __restrict__ nv1,
                            const float* __restrict__ nv2,
                            const float* __restrict__ g1w, const float* __restrict__ g1b,
                            const float* __restrict__ g2w, const float* __restrict__ g2b,
                            const float* __restrict__ n1w, const float* __restrict__ n1b,
                            const float* __restrict__ n2w, const float* __restrict__ n2b,
                            float* __restrict__ xp1, float* __restrict__ xp2) {
    int t = blockIdx.x * blockDim.x + threadIdx.x;
    if (t >= kSEQ) return;
    int n = (t / kM) % kN;
    const float* xv = x + (long)t * 64;
    float a1 = g1b[0], a2 = g2b[0];
    for (int d = 0; d < 64; ++d) {
        a1 += xv[d] * g1w[d];
        a2 += xv[d] * g2w[d];
    }
    for (int v = 0; v < kND; ++v) {
        a1 += nv1[n * kND + v] * g1w[64 + v];
        a2 += nv2[v * kN + n] * g2w[64 + v];
    }
    float gg1 = fmaxf(tanhf(a1), 0.f);
    float gg2 = fmaxf(tanhf(a2), 0.f);
    for (int v = 0; v < kND; ++v) {
        float p1 = n1b[v], p2 = n2b[v];
        for (int d = 0; d < 64; ++d) {
            p1 += xv[d] * n1w[v * 64 + d];
            p2 += xv[d] * n2w[v * 64 + d];
        }
        xp1[(long)t * kND + v] = gg1 * p1;
        xp2[(long)t * kND + v] = gg2 * p2;
    }
}

// ---- adaptive adjacency: adp[b,m] = softmax_row(relu(nv1' @ nv2')) ----
__global__ void adp_kernel(const float* __restrict__ xp1, const float* __restrict__ xp2,
                           const float* __restrict__ nv1, const float* __restrict__ nv2,
                           float* __restrict__ adp) {
    const int bm = blockIdx.x;
    const int b = bm / kM, mp = bm % kM;
    __shared__ float sc[kN][kN];
    __shared__ float n1[kN][kND], n2[kND][kN];
    const int tid = threadIdx.x;
    for (int i = tid; i < kN * kND; i += blockDim.x) {
        int nn = i / kND, v = i % kND;
        long toff = (((long)b * kN + nn) * kM + mp) * kND + v;
        n1[nn][v] = nv1[nn * kND + v] + xp1[toff];
        n2[v][nn] = nv2[v * kN + nn] + xp2[toff];
    }
    __syncthreads();
    for (int i = tid; i < kN * kN; i += blockDim.x) {
        int r = i / kN, c = i % kN;
        float s = 0.f;
        for (int v = 0; v < kND; ++v) s += n1[r][v] * n2[v][c];
        sc[r][c] = fmaxf(s, 0.f);
    }
    __syncthreads();
    if (tid < kN) {
        int r = tid;
        float mx = sc[r][0];
        for (int c = 1; c < kN; ++c) mx = fmaxf(mx, sc[r][c]);
        float ss = 0.f;
        for (int c = 0; c < kN; ++c) {
            float e = __expf(sc[r][c] - mx);
            sc[r][c] = e;
            ss += e;
        }
        float inv = 1.f / ss;
        long base = (((long)b * kM + mp) * kN + r) * kN;
        for (int c = 0; c < kN; ++c) adp[base + c] = sc[r][c] * inv;
    }
}

// ---- copy x into hcat channels [0,64) (token-major, stride 192) ----
__global__ void copy_hcat_kernel(const float* __restrict__ x, float* __restrict__ hcat) {
    int i = blockIdx.x * blockDim.x + threadIdx.x;
    if (i >= kSEQ * 64) return;
    int t = i >> 6, d = i & 63;
    hcat[(long)t * 192 + d] = x[i];
}

// ---- one graph-conv hop: out[b,v,m,f] = sum_n in[b,n,m,f] * adp[b,m,n,v] ----
__global__ void hop_kernel(const float* __restrict__ hcat_src, int src_off,
                           const float* __restrict__ adp, float* __restrict__ hcat_dst,
                           int dst_off) {
    const int bm = blockIdx.x;
    const int b = bm / kM, mp = bm % kM;
    __shared__ float sx[kN][64];
    __shared__ float sa[kN][kN];
    const int tid = threadIdx.x;
    for (int i = tid; i < kN * 64; i += blockDim.x) {
        int nn = i >> 6, f = i & 63;
        sx[nn][f] = hcat_src[(((long)b * kN + nn) * kM + mp) * 192 + src_off + f];
    }
    for (int i = tid; i < kN * kN; i += blockDim.x)
        sa[i / kN][i % kN] = adp[((long)b * kM + mp) * kN * kN + i];
    __syncthreads();
    for (int i = tid; i < kN * 64; i += blockDim.x) {
        int v = i >> 6, f = i & 63;
        float s = 0.f;
        for (int nn = 0; nn < kN; ++nn) s += sx[nn][f] * sa[nn][v];
        hcat_dst[(((long)b * kN + v) * kM + mp) * 192 + dst_off + f] = s;
    }
}

// ---- residual add: y += x ----
__global__ void axpy_kernel(const float* __restrict__ x, float* __restrict__ y) {
    int i = blockIdx.x * blockDim.x + threadIdx.x;
    if (i < kSEQ * 64) y[i] += x[i];
}

// ---- decoder token build: hd = [hx(b,n), te(tp(b,l)), 0-pad] (96 wide) ----
__global__ void hd_kernel(const float* __restrict__ hx, const float* __restrict__ tsp,
                          const float* __restrict__ sw, const float* __restrict__ sb,
                          const float* __restrict__ pw, const float* __restrict__ pb,
                          float* __restrict__ hd) {
    int t = blockIdx.x * blockDim.x + threadIdx.x;
    if (t >= kDTOK) return;
    int l = t % kLP;
    int n = (t / kLP) % kN;
    int b = t / (kLP * kN);
    float* o = hd + (long)t * 96;
    const float* h = hx + ((long)b * kN + n) * 64;
    for (int d = 0; d < 64; ++d) o[d] = h[d];
    float tp = tsp[b * kLP + l];
    o[64] = tp * sw[0] + sb[0];
    for (int i = 0; i < kTE - 1; ++i) o[65 + i] = __sinf(tp * pw[i] + pb[i]);
    for (int d = 74; d < 96; ++d) o[d] = 0.f;  // zero pad -> guard-free K=96 GEMM
}

// ---- final projection to scalar + transpose into (B, L_PRED, N) ----
__global__ void dec3_kernel(const float* __restrict__ d2, const float* __restrict__ w,
                            const float* __restrict__ bsc, float* __restrict__ out) {
    int t = blockIdx.x * blockDim.x + threadIdx.x;
    if (t >= kDTOK) return;
    float s = bsc[0];
    for (int d = 0; d < 64; ++d) s += d2[(long)t * 64 + d] * w[d];
    int l = t % kLP;
    int n = (t / kLP) % kN;
    int b = t / (kLP * kN);
    out[((long)b * kLP + l) * kN + n] = s;
}

// -------- input pointer indices (setup_inputs dict order, recursive) --------
enum {
    IN_TSP = 0, IN_X, IN_TTS, IN_MASK,
    P_TESW, P_TESB, P_TEPW, P_TEPB,
    P_FG1W, P_FG1B, P_FG2W, P_FG2B, P_FG3W, P_FG3B,
    P_TBIAS, P_NV1, P_NV2,
    P_LAYER0 = 17,   // 22 entries per layer, 2 layers
    P_AGGW = 17 + 2 * 22, P_AGGB, P_DEC1W, P_DEC1B, P_DEC2W, P_DEC2B, P_DEC3W, P_DEC3B
};
enum {
    L_IN_W = 0, L_IN_B, L_OUT_W, L_OUT_B, L_FF1_W, L_FF1_B, L_FF2_W, L_FF2_B,
    L_LN1_G, L_LN1_B, L_LN2_G, L_LN2_B, L_NVL1_W, L_NVL1_B, L_NVL2_W, L_NVL2_B,
    L_G1_W, L_G1_B, L_G2_W, L_G2_B, L_GC_W, L_GC_B
};

extern "C" void kernel_launch(void* const* d_in, const int* in_sizes, int n_in,
                              void* d_out, int out_size, void* d_ws, size_t ws_size,
                              hipStream_t stream) {
    (void)in_sizes; (void)n_in; (void)out_size; (void)ws_size;
    auto F = [&](int i) { return (const float*)d_in[i]; };

    // ---- workspace bump allocator (all sizes multiples of 8 floats ->
    //      every pointer stays 32B aligned for float4 / v16h loads) ----
    float* w = (float*)d_ws;
    auto alloc = [&](size_t nflt) { float* p = w; w += nflt; return p; };
    auto alloch = [&](size_t nhalf) { return (_Float16*)alloc(nhalf / 2); };

    float* xA   = alloc((size_t)kSEQ * 64);
    float* xB   = alloc((size_t)kSEQ * 64);
    float* xpe  = alloc((size_t)kSEQ * 64);
    float* qkv  = alloc((size_t)kSEQ * 192);
    float* obuf = alloc((size_t)kSEQ * 64);
    float* proj = alloc((size_t)kSEQ * 64);
    float* e1   = alloc((size_t)kSEQ * 64);
    float* ffh  = alloc((size_t)kSEQ * kDFF);
    float* ffo  = alloc((size_t)kSEQ * 64);
    float* enc  = alloc((size_t)kSEQ * 64);
    float* xp1  = alloc((size_t)kSEQ * kND);
    float* xp2  = alloc((size_t)kSEQ * kND);
    float* adp  = alloc((size_t)kB * kM * kN * kN);
    float* hcat = alloc((size_t)kSEQ * 192);
    float* hx   = alloc((size_t)kB * kN * 64);
    float* hd   = alloc((size_t)kDTOK * 96);
    float* d1   = alloc((size_t)kDTOK * 64);
    float* d2   = alloc((size_t)kDTOK * 64);

    // f16 (padded) weight copies
    _Float16* fg3h  = alloch(704 * 64);
    _Float16* aggh  = alloch(64 * 1024);
    _Float16* dec1h = alloch(64 * 96);
    _Float16* dec2h = alloch(64 * 64);
    _Float16* inh[kNLAYER], *outh[kNLAYER], *ff1h[kNLAYER], *ff2h[kNLAYER], *gch[kNLAYER];
    for (int li = 0; li < kNLAYER; ++li) {
        inh[li]  = alloch(192 * 64);
        outh[li] = alloch(64 * 64);
        ff1h[li] = alloch((size_t)kDFF * 64);
        ff2h[li] = alloch((size_t)64 * kDFF);
        gch[li]  = alloch(64 * 192);
    }

    // ---- weight prep (f32 -> padded f16) ----
    auto prep = [&](const float* src, _Float16* dst, int rs, int ks, int rd, int kd) {
        pad_cvt_kernel<<<(rd * kd + 255) / 256, 256, 0, stream>>>(src, dst, rs, ks, rd, kd);
    };
    prep(F(P_FG3W), fg3h, 693, 63, 704, 64);
    prep(F(P_AGGW), aggh, 64, 1024, 64, 1024);
    prep(F(P_DEC1W), dec1h, 64, 74, 64, 96);
    prep(F(P_DEC2W), dec2h, 64, 64, 64, 64);
    for (int li = 0; li < kNLAYER; ++li) {
        const int p0 = P_LAYER0 + li * 22;
        prep(F(p0 + L_IN_W),  inh[li],  192, 64, 192, 64);
        prep(F(p0 + L_OUT_W), outh[li], 64, 64, 64, 64);
        prep(F(p0 + L_FF1_W), ff1h[li], kDFF, 64, kDFF, 64);
        prep(F(p0 + L_FF2_W), ff2h[li], 64, kDFF, 64, kDFF);
        prep(F(p0 + L_GC_W),  gch[li],  64, 192, 64, 192);
    }

    // ---- stage 1: fused TTCN (TE + fg1 + fg2 + WMMA fg3 + softmax) ----
    ttcn_kernel<<<kSEQ, 256, 0, stream>>>(
        F(IN_X), F(IN_TTS), F(IN_MASK), F(P_TESW), F(P_TESB), F(P_TEPW), F(P_TEPB),
        F(P_FG1W), F(P_FG1B), F(P_FG2W), F(P_FG2B), fg3h, F(P_FG3B), F(P_TBIAS), xA);

    float* xcur = xA;
    float* xnext = xB;

    for (int li = 0; li < kNLAYER; ++li) {
        const int p0 = P_LAYER0 + li * 22;
        // ---- transformer encoder (post-LN) ----
        addpe_kernel<<<(kSEQ * 64 + 255) / 256, 256, 0, stream>>>(xcur, xpe);
        gemm_nt<0, 4, 2><<<dim3(kSEQ / 64, 192 / 32), 32, 0, stream>>>(
            xpe, inh[li], F(p0 + L_IN_B), qkv, 192, 64);
        attn_kernel<<<kB * kN * kNHEAD, 256, 0, stream>>>(qkv, obuf);
        gemm_nt<0, 4, 2><<<dim3(kSEQ / 64, 2), 32, 0, stream>>>(
            obuf, outh[li], F(p0 + L_OUT_B), proj, 64, 64);
        addln_kernel<<<kSEQ, 64, 0, stream>>>(xpe, proj, F(p0 + L_LN1_G), F(p0 + L_LN1_B), e1);
        gemm_nt<1, 4, 2><<<dim3(kSEQ / 64, kDFF / 32), 32, 0, stream>>>(
            e1, ff1h[li], F(p0 + L_FF1_B), ffh, kDFF, 64);
        gemm_nt<0, 4, 2><<<dim3(kSEQ / 64, 2), 32, 0, stream>>>(
            ffh, ff2h[li], F(p0 + L_FF2_B), ffo, 64, kDFF);
        addln_kernel<<<kSEQ, 64, 0, stream>>>(e1, ffo, F(p0 + L_LN2_G), F(p0 + L_LN2_B), enc);
        // ---- adaptive graph construction ----
        gate_kernel<<<(kSEQ + 255) / 256, 256, 0, stream>>>(
            enc, F(P_NV1), F(P_NV2), F(p0 + L_G1_W), F(p0 + L_G1_B), F(p0 + L_G2_W),
            F(p0 + L_G2_B), F(p0 + L_NVL1_W), F(p0 + L_NVL1_B), F(p0 + L_NVL2_W),
            F(p0 + L_NVL2_B), xp1, xp2);
        adp_kernel<<<kB * kM, 256, 0, stream>>>(xp1, xp2, F(P_NV1), F(P_NV2), adp);
        // ---- multi-hop graph conv -> hcat (token-major, 192 channels) ----
        copy_hcat_kernel<<<(kSEQ * 64 + 255) / 256, 256, 0, stream>>>(enc, hcat);
        hop_kernel<<<kB * kM, 256, 0, stream>>>(hcat, 0, adp, hcat, 64);
        hop_kernel<<<kB * kM, 256, 0, stream>>>(hcat, 64, adp, hcat, 128);
        gemm_nt<1, 4, 2><<<dim3(kSEQ / 64, 2), 32, 0, stream>>>(
            hcat, gch[li], F(p0 + L_GC_B), xnext, 64, 192);
        if (li > 0)
            axpy_kernel<<<(kSEQ * 64 + 255) / 256, 256, 0, stream>>>(xcur, xnext);
        float* tmp = xcur; xcur = xnext; xnext = tmp;
    }

    // ---- temporal aggregation + decoder ----
    gemm_nt<0, 1, 2><<<dim3((kB * kN) / 16, 2), 32, 0, stream>>>(
        xcur, aggh, F(P_AGGB), hx, 64, kM * kHID);
    hd_kernel<<<(kDTOK + 255) / 256, 256, 0, stream>>>(
        hx, F(IN_TSP), F(P_TESW), F(P_TESB), F(P_TEPW), F(P_TEPB), hd);
    gemm_nt<1, 4, 2><<<dim3(kDTOK / 64, 2), 32, 0, stream>>>(
        hd, dec1h, F(P_DEC1B), d1, 64, 96);
    gemm_nt<1, 4, 2><<<dim3(kDTOK / 64, 2), 32, 0, stream>>>(
        d1, dec2h, F(P_DEC2B), d2, 64, 64);
    dec3_kernel<<<(kDTOK + 255) / 256, 256, 0, stream>>>(
        d2, F(P_DEC3W), F(P_DEC3B), (float*)d_out);
}
